// Generator_74079595921631
// MI455X (gfx1250) — matrix operand order repaired and checked by reading, not compiled, not run on hardware
//
#include <hip/hip_runtime.h>
#include <math.h>

typedef __attribute__((ext_vector_type(16))) __bf16 v16bf;
typedef __attribute__((ext_vector_type(8)))  float  v8f;

#define WMMA_BF16(A_, B_, C_) \
  __builtin_amdgcn_wmma_f32_16x16x32_bf16(false, (A_), false, (B_), (short)0, (C_), false, false)

// Problem constants
constexpr int   Bsz = 4096, Tn = 128, IN = 32, Hdim = 128, OUTD = 32, NS = 20;
constexpr float DTc = 0.05f;

// Tiling
constexpr int THREADS = 128;     // 4 waves (wave32)
constexpr int M_TILE  = 64;      // 16 rows per wave
constexpr int KSTR    = 136;     // padded LDS stride (bank-conflict-free b128 loads)
constexpr int KSTR_I  = 40;      // padded stride for W_init (K=32)

// LDS element counts (bf16)
constexpr int    W128_EL = 128 * KSTR;
constexpr int    WI_EL   = 128 * KSTR_I;
constexpr int    WD_EL   = 32 * KSTR;
constexpr int    ZB_EL   = M_TILE * KSTR;
constexpr size_t BF16_EL = 6ull * W128_EL + WI_EL + WD_EL + 3ull * ZB_EL;
constexpr size_t F32_EL  = 9 * 128 + 32 + 128;      // biases + t-channel rows + b_dec + times
constexpr size_t SMEM_BYTES = BF16_EL * 2 + F32_EL * 4;   // ~285 KB < 320 KB/WGP
static_assert(SMEM_BYTES < 320 * 1024, "LDS budget");

__device__ __forceinline__ v16bf load_a_frag(const __bf16* buf, int m0, int k0) {
  int lane = threadIdx.x & 31, half = lane >> 4, r = lane & 15;
  const __bf16* p = buf + (m0 + r) * KSTR + k0 + half * 8;
  union { uint4 u[2]; v16bf v; } un;
  un.u[0] = *(const uint4*)p;          // K = k0 + 8*half .. +7
  un.u[1] = *(const uint4*)(p + 16);   // K = k0 + 16 + 8*half .. +7
  return un.v;
}

__device__ __forceinline__ v16bf load_b_frag(const __bf16* wt, int n0, int k0, int stride) {
  int lane = threadIdx.x & 31, half = lane >> 4, n = lane & 15;
  const __bf16* p = wt + (n0 + n) * stride + k0 + half * 8;
  union { uint4 u[2]; v16bf v; } un;
  un.u[0] = *(const uint4*)p;          // K = k0 + 8*half .. +7
  un.u[1] = *(const uint4*)(p + 16);   // K = k0 + 16 + 8*half .. +7
  return un.v;
}

// One dense layer: [16 x 128] (bf16 in abuf) @ [128 x 128] (Wt transposed) -> 8 f32 C tiles
__device__ __forceinline__ void gemm128(const __bf16* abuf, const __bf16* wt, int m0, v8f acc[8]) {
  v16bf A0 = load_a_frag(abuf, m0, 0);
  v16bf A1 = load_a_frag(abuf, m0, 32);
  v16bf A2 = load_a_frag(abuf, m0, 64);
  v16bf A3 = load_a_frag(abuf, m0, 96);
#pragma unroll
  for (int n = 0; n < 8; ++n) {
    v8f c = {0.f, 0.f, 0.f, 0.f, 0.f, 0.f, 0.f, 0.f};
    c = WMMA_BF16(A0, load_b_frag(wt, n * 16, 0, KSTR), c);
    c = WMMA_BF16(A1, load_b_frag(wt, n * 16, 32, KSTR), c);
    c = WMMA_BF16(A2, load_b_frag(wt, n * 16, 64, KSTR), c);
    c = WMMA_BF16(A3, load_b_frag(wt, n * 16, 96, KSTR), c);
    acc[n] = c;
  }
}

// lipswish(acc + bias [+ t*wlast]) -> bf16 LDS buffer (C layout -> row major)
__device__ __forceinline__ void act_store(__bf16* dst, const v8f acc[8], const float* bias,
                                          const float* wlast, float t, int m0, int half, int c16) {
#pragma unroll
  for (int n = 0; n < 8; ++n) {
    int col = n * 16 + c16;
    float bb = bias[col] + (wlast ? t * wlast[col] : 0.0f);
#pragma unroll
    for (int j = 0; j < 8; ++j) {
      float x = acc[n][j] + bb;
      float y = 0.909f * x * (1.0f / (1.0f + __expf(-x)));
      dst[(m0 + j + 8 * half) * KSTR + col] = (__bf16)y;
    }
  }
}

// u = z @ W_dec + b_dec  (N = 32 -> 2 tiles)
__device__ __forceinline__ void compute_u(const __bf16* zb, const __bf16* wtD, const float* bDecS,
                                          int m0, int c16, v8f u[2]) {
  v16bf A0 = load_a_frag(zb, m0, 0);
  v16bf A1 = load_a_frag(zb, m0, 32);
  v16bf A2 = load_a_frag(zb, m0, 64);
  v16bf A3 = load_a_frag(zb, m0, 96);
#pragma unroll
  for (int nt = 0; nt < 2; ++nt) {
    v8f c = {0.f, 0.f, 0.f, 0.f, 0.f, 0.f, 0.f, 0.f};
    c = WMMA_BF16(A0, load_b_frag(wtD, nt * 16, 0, KSTR), c);
    c = WMMA_BF16(A1, load_b_frag(wtD, nt * 16, 32, KSTR), c);
    c = WMMA_BF16(A2, load_b_frag(wtD, nt * 16, 64, KSTR), c);
    c = WMMA_BF16(A3, load_b_frag(wtD, nt * 16, 96, KSTR), c);
    float bb = bDecS[nt * 16 + c16];
#pragma unroll
    for (int j = 0; j < 8; ++j) u[nt][j] = c[j] + bb;
  }
}

__global__ __launch_bounds__(THREADS, 1) void sde_fused(
    const float* __restrict__ coeffs, const float* __restrict__ times,
    const float* __restrict__ noise,
    const float* __restrict__ W_init, const float* __restrict__ b_init,
    const float* __restrict__ W_dec, const float* __restrict__ b_dec,
    const float* __restrict__ fW0, const float* __restrict__ fb0,
    const float* __restrict__ fW1, const float* __restrict__ fb1,
    const float* __restrict__ fW2, const float* __restrict__ fb2,
    const float* __restrict__ gW0, const float* __restrict__ gb0,
    const float* __restrict__ gW1, const float* __restrict__ gb1,
    const float* __restrict__ gW2, const float* __restrict__ gb2,
    float* __restrict__ out) {
  extern __shared__ __align__(16) char smem[];
  __bf16* WtF0 = (__bf16*)smem;
  __bf16* WtF1 = WtF0 + W128_EL;
  __bf16* WtF2 = WtF1 + W128_EL;
  __bf16* WtG0 = WtF2 + W128_EL;
  __bf16* WtG1 = WtG0 + W128_EL;
  __bf16* WtG2 = WtG1 + W128_EL;
  __bf16* WtI  = WtG2 + W128_EL;
  __bf16* WtD  = WtI + WI_EL;
  __bf16* zbf  = WtD + WD_EL;
  __bf16* hA   = zbf + ZB_EL;
  __bf16* hB   = hA + ZB_EL;
  float* bInitS = (float*)(hB + ZB_EL);
  float* bF0S = bInitS + 128;
  float* wF0L = bF0S + 128;
  float* bF1S = wF0L + 128;
  float* bF2S = bF1S + 128;
  float* bG0S = bF2S + 128;
  float* wG0L = bG0S + 128;
  float* bG1S = wG0L + 128;
  float* bG2S = bG1S + 128;
  float* bDecS = bG2S + 128;
  float* timesS = bDecS + 32;

  const int tid = threadIdx.x;

  // ---- Stage weights (transposed, bf16) + biases into LDS, once per block ----
  {
    const float* src[6] = {fW0, fW1, fW2, gW0, gW1, gW2};
    __bf16* dst[6] = {WtF0, WtF1, WtF2, WtG0, WtG1, WtG2};
#pragma unroll 1
    for (int m = 0; m < 6; ++m) {
      const float* W = src[m];
      __bf16* wt = dst[m];
      for (int idx = tid; idx < 128 * 128; idx += THREADS) {
        int k = idx >> 7, n = idx & 127;
        wt[n * KSTR + k] = (__bf16)W[idx];        // coalesced read, transposed write
      }
    }
    for (int i = tid; i < 128; i += THREADS) {
      bInitS[i] = b_init[i];
      bF0S[i] = fb0[i];  wF0L[i] = fW0[128 * 128 + i];   // t-channel row of [129,128]
      bF1S[i] = fb1[i];  bF2S[i] = fb2[i];
      bG0S[i] = gb0[i];  wG0L[i] = gW0[128 * 128 + i];
      bG1S[i] = gb1[i];  bG2S[i] = gb2[i];
      timesS[i] = times[i];
    }
    for (int i = tid; i < 32; i += THREADS) bDecS[i] = b_dec[i];
    for (int idx = tid; idx < 32 * 128; idx += THREADS) {   // W_init [32,128]
      int k = idx >> 7, n = idx & 127;
      WtI[n * KSTR_I + k] = (__bf16)W_init[idx];
    }
    for (int idx = tid; idx < 128 * 32; idx += THREADS) {   // W_dec [128,32]
      int k = idx >> 5, n = idx & 31;
      WtD[n * KSTR + k] = (__bf16)W_dec[idx];
    }
  }
  __syncthreads();

  const int lane = tid & 31;
  const int wave = tid >> 5;
  const int half = lane >> 4;
  const int c16  = lane & 15;
  const int m0   = wave * 16;                         // block-local row base
  const int browBase = blockIdx.x * M_TILE + m0;      // global batch row base

  v8f z[8];   // state, C layout: z[n][j] = z[row=j+8*half, col=16n+c16]

  // ---- z0 = coeffs[:,0,:] @ W_init + b_init ----
  {
    union { uint4 u[2]; v16bf v; } Ai;
    const float* cp = coeffs + (size_t)(browBase + c16) * (Tn * IN);
#pragma unroll
    for (int q = 0; q < 8; ++q) {
      Ai.v[q]     = (__bf16)cp[half * 8 + q];
      Ai.v[8 + q] = (__bf16)cp[16 + half * 8 + q];
    }
#pragma unroll
    for (int n = 0; n < 8; ++n) {
      v8f c = {0.f, 0.f, 0.f, 0.f, 0.f, 0.f, 0.f, 0.f};
      c = WMMA_BF16(Ai.v, load_b_frag(WtI, n * 16, 0, KSTR_I), c);
      float bb = bInitS[n * 16 + c16];
#pragma unroll
      for (int j = 0; j < 8; ++j) z[n][j] = c[j] + bb;
    }
  }
#pragma unroll
  for (int n = 0; n < 8; ++n) {
    int col = n * 16 + c16;
#pragma unroll
    for (int j = 0; j < 8; ++j)
      zbf[(m0 + j + 8 * half) * KSTR + col] = (__bf16)z[n][j];
  }
  asm volatile("" ::: "memory");

  v8f uprev[2];
  compute_u(zbf, WtD, bDecS, m0, c16, uprev);

  // ---- 20 Euler steps, fully fused ----
#pragma unroll 1
  for (int k = 0; k < NS; ++k) {
    float t = k * DTc;
    v8f acc[8];

    // drift MLP
    asm volatile("" ::: "memory");
    gemm128(zbf, WtF0, m0, acc);
    act_store(hA, acc, bF0S, wF0L, t, m0, half, c16);
    asm volatile("" ::: "memory");
    gemm128(hA, WtF1, m0, acc);
    act_store(hB, acc, bF1S, nullptr, 0.f, m0, half, c16);
    asm volatile("" ::: "memory");
    v8f drift[8];
    gemm128(hB, WtF2, m0, drift);

    // diffusion MLP
    gemm128(zbf, WtG0, m0, acc);
    act_store(hA, acc, bG0S, wG0L, t, m0, half, c16);
    asm volatile("" ::: "memory");
    gemm128(hA, WtG1, m0, acc);
    act_store(hB, acc, bG1S, nullptr, 0.f, m0, half, c16);
    asm volatile("" ::: "memory");
    v8f gacc[8];
    gemm128(hB, WtG2, m0, gacc);

    // z += DT * (drift + tanh(g) * eps)
    const float* nz = noise + (size_t)k * Bsz * Hdim;
#pragma unroll
    for (int n = 0; n < 8; ++n) {
      int col = n * 16 + c16;
      float bf2 = bF2S[col], bg2 = bG2S[col];
#pragma unroll
      for (int j = 0; j < 8; ++j) {
        int brow = browBase + j + 8 * half;
        float eps = nz[(size_t)brow * Hdim + col];
        float d = drift[n][j] + bf2;
        float g = tanhf(gacc[n][j] + bg2);
        z[n][j] += DTc * (d + g * eps);
      }
    }
#pragma unroll
    for (int n = 0; n < 8; ++n) {
      int col = n * 16 + c16;
#pragma unroll
      for (int j = 0; j < 8; ++j)
        zbf[(m0 + j + 8 * half) * KSTR + col] = (__bf16)z[n][j];
    }
    asm volatile("" ::: "memory");

    // u_{k+1} = z @ W_dec + b_dec; emit all output times interpolating [k, k+1]
    v8f ucur[2];
    compute_u(zbf, WtD, bDecS, m0, c16, ucur);

#pragma unroll 1
    for (int i = 0; i < Tn; ++i) {
      float pos = timesS[i] * (1.0f / DTc);
      int idx = (int)floorf(pos);
      idx = idx < 0 ? 0 : (idx > NS - 1 ? NS - 1 : idx);
      if (idx != k) continue;
      float f = pos - (float)idx;
#pragma unroll
      for (int nt = 0; nt < 2; ++nt) {
        int col = nt * 16 + c16;
#pragma unroll
        for (int j = 0; j < 8; ++j) {
          float val = (1.0f - f) * uprev[nt][j] + f * ucur[nt][j];
          out[(size_t)(browBase + j + 8 * half) * (Tn * OUTD) + (size_t)i * OUTD + col] = val;
        }
      }
    }
    uprev[0] = ucur[0];
    uprev[1] = ucur[1];
  }
}

extern "C" void kernel_launch(void* const* d_in, const int* in_sizes, int n_in,
                              void* d_out, int out_size, void* d_ws, size_t ws_size,
                              hipStream_t stream) {
  (void)in_sizes; (void)n_in; (void)out_size; (void)d_ws; (void)ws_size;
  const float* coeffs = (const float*)d_in[0];
  const float* times  = (const float*)d_in[1];
  const float* noise  = (const float*)d_in[2];
  const float* W_init = (const float*)d_in[3];
  const float* b_init = (const float*)d_in[4];
  const float* W_dec  = (const float*)d_in[5];
  const float* b_dec  = (const float*)d_in[6];
  const float* fW0 = (const float*)d_in[7];
  const float* fb0 = (const float*)d_in[8];
  const float* fW1 = (const float*)d_in[9];
  const float* fb1 = (const float*)d_in[10];
  const float* fW2 = (const float*)d_in[11];
  const float* fb2 = (const float*)d_in[12];
  const float* gW0 = (const float*)d_in[13];
  const float* gb0 = (const float*)d_in[14];
  const float* gW1 = (const float*)d_in[15];
  const float* gb1 = (const float*)d_in[16];
  const float* gW2 = (const float*)d_in[17];
  const float* gb2 = (const float*)d_in[18];
  float* out = (float*)d_out;

  (void)hipFuncSetAttribute((const void*)sde_fused,
                            hipFuncAttributeMaxDynamicSharedMemorySize,
                            (int)SMEM_BYTES);
  sde_fused<<<Bsz / M_TILE, THREADS, SMEM_BYTES, stream>>>(
      coeffs, times, noise, W_init, b_init, W_dec, b_dec,
      fW0, fb0, fW1, fb1, fW2, fb2, gW0, gb0, gW1, gb1, gW2, gb2, out);
}
